// GatedTransformerXL_73126113181745
// MI455X (gfx1250) — matrix operand order repaired and checked by reading, not compiled, not run on hardware
//
#include <hip/hip_runtime.h>

// ---------------------------------------------------------------------------
// Gated Transformer-XL forward for MI455X (gfx1250).
// All GEMMs run through v_wmma_f32_16x16x32_bf16 (fp32 accumulate) with
// double-buffered LDS tiles fed entirely by GLOBAL_LOAD_ASYNC_TO_LDS_B128
// (ASYNCcnt). Weights are pre-packed transposed (N x K, bf16) so every B
// operand is k-contiguous: one LDS layout, pure b128 DS traffic, no gathers.
// Roofline: ~1 TFLOP of GEMM vs ~0.5 GB unique traffic => compute bound.
// ---------------------------------------------------------------------------

typedef __attribute__((ext_vector_type(16))) __bf16 v16bf;
typedef __attribute__((ext_vector_type(8)))  float  v8f;
typedef int v4i_async __attribute__((vector_size(16)));   // async b128 unit

#define Lc  4
#define Sc  512
#define Mc  512
#define Bc  8
#define Dc  1024
#define Hc  16
#define INc 128
#define Fc  1024   // S + M
#define HDc 64
#define SBc 4096   // S*B
#define FBc 8192   // F*B

#if __has_builtin(__builtin_amdgcn_global_load_async_to_lds_b128) && \
    __has_builtin(__builtin_amdgcn_s_wait_asynccnt)
#define ASYNC_LDS 1
#else
#define ASYNC_LDS 0
#endif

#define GLOBAL_AS __attribute__((address_space(1)))
#define LDS_AS    __attribute__((address_space(3)))
#define ASY_G(p)  ((GLOBAL_AS v4i_async*)(p))
#define ASY_L(p)  ((LDS_AS    v4i_async*)(p))

// ----------------------------- WMMA GEMM -----------------------------------
// C[b][m][n] = ACT( sum_k A[b][m][k]*B[b][k][n] + bias[n] + addend[b][m][n] )
// A is k-contiguous (lda = row stride). B is ALWAYS k-contiguous:
// B[k][n] = Bm[offB + n*ldbN + k]  (weights are pre-packed transposed).
// Batch off = (z/bdiv)*S1 + (z%bdiv)*S2 per tensor.
// Optional CbT: transposed bf16 output CbT[(m%8)*ctS1 + (m/8)*ctS2 + n*ldcT]
// (used to produce V in k-contiguous form for the A@V GEMM).
#define BM 128
#define BK 32

template <int BNt, int ACT>
__global__ __launch_bounds__(256)
void k_gemm(const __bf16* __restrict__ A, long long aS1, long long aS2, int lda,
            const __bf16* __restrict__ Bm, long long bS1, long long bS2, int ldbN,
            float* __restrict__ Cf, __bf16* __restrict__ Cb,
            __bf16* __restrict__ CbT, long long ctS1, long long ctS2, int ldcT,
            long long cS1, long long cS2, int ldc,
            const float* __restrict__ bias, const float* __restrict__ addend,
            int K, int bdiv)
{
    constexpr int FN  = BNt / 32;      // n-fragments per wave (2 waves span BNt)
    constexpr int BCH = BNt / 64;      // B-staging b128 units per thread

    __shared__ __bf16 Asm[2][BM][BK + 8];    // 80B rows -> 16B aligned
    __shared__ __bf16 BsT[2][BNt][BK + 8];   // transposed: k contiguous per n

    const int t     = threadIdx.x;
    const int lane  = t & 31;
    const int w     = t >> 5;
    const int waveM = w & 3;           // 4 waves along M: 4*32 = 128
    const int waveN = w >> 2;          // 2 waves along N: 2*(BNt/2)
    const int z     = blockIdx.z;
    const long long offA = (long long)(z / bdiv) * aS1 + (long long)(z % bdiv) * aS2;
    const long long offB = (long long)(z / bdiv) * bS1 + (long long)(z % bdiv) * bS2;
    const long long offC = (long long)(z / bdiv) * cS1 + (long long)(z % bdiv) * cS2;
    const int m0 = blockIdx.y * BM;
    const int n0 = blockIdx.x * BNt;

    v8f acc[2][FN];
    const v8f vzero = {0.f, 0.f, 0.f, 0.f, 0.f, 0.f, 0.f, 0.f};
    #pragma unroll
    for (int fm = 0; fm < 2; ++fm)
        #pragma unroll
        for (int fn = 0; fn < FN; ++fn) acc[fm][fn] = vzero;

    const int arow = t >> 1, akoff = (t & 1) * 16;   // A: 2 thr/row, 16 bf16 each

    auto stage = [&](int k0, int buf) {
        // ---- A tile 128x32 bf16 (k-contiguous) ----
        const __bf16* ap = A + offA + (long long)(m0 + arow) * lda + (k0 + akoff);
        __bf16* lp = &Asm[buf][arow][akoff];
#if ASYNC_LDS
        __builtin_amdgcn_global_load_async_to_lds_b128(ASY_G(ap), ASY_L(lp), 0, 0);
        __builtin_amdgcn_global_load_async_to_lds_b128(ASY_G(ap + 8), ASY_L(lp + 8), 0, 0);
#else
        ((uint4*)lp)[0] = ((const uint4*)ap)[0];
        ((uint4*)lp)[1] = ((const uint4*)ap)[1];
#endif
        // ---- B tile BNt x 32 bf16 (k-contiguous) ----
        #pragma unroll
        for (int c = 0; c < BCH; ++c) {
            const int uidx = t + c * 256;       // b128 unit: 4 per column
            const int bn   = uidx >> 2;
            const int koff = (uidx & 3) * 8;
            const __bf16* bp = Bm + offB + (long long)(n0 + bn) * ldbN + (k0 + koff);
            __bf16* blp = &BsT[buf][bn][koff];
#if ASYNC_LDS
            __builtin_amdgcn_global_load_async_to_lds_b128(ASY_G(bp), ASY_L(blp), 0, 0);
#else
            *(uint4*)blp = *(const uint4*)bp;
#endif
        }
    };

    const int nT = K / BK;
    stage(0, 0);
    for (int ti = 0; ti < nT; ++ti) {
        const bool hasNext = (ti + 1) < nT;
        if (hasNext) stage((ti + 1) * BK, (ti + 1) & 1);   // prefetch next tile
#if ASYNC_LDS
        // allow only next tile's async copies (2 A + BCH B) to stay outstanding
        if (hasNext) __builtin_amdgcn_s_wait_asynccnt(2 + BCH);
        else         __builtin_amdgcn_s_wait_asynccnt(0);
#endif
        __syncthreads();

        const __bf16 (*As)[BK + 8] = Asm[ti & 1];
        const __bf16 (*Bs)[BK + 8] = BsT[ti & 1];

        // ---- fragments per ISA layout (16-bit A 16x32 / B 32x16, wave32) ----
        const int ml = lane & 15;
        const int kh = (lane >> 4) << 3;   // A: lane half -> k sub-block of 8
        const int kb = (lane >> 4) << 4;   // B: lane half -> k sub-block of 8 (used as kb>>1)
        v16bf af[2], bfr[FN];
        #pragma unroll
        for (int fm = 0; fm < 2; ++fm) {
            const int m = waveM * 32 + fm * 16 + ml;
            #pragma unroll
            for (int e = 0; e < 16; ++e)
                af[fm][e] = As[m][((e >> 3) << 4) + kh + (e & 7)];
        }
        #pragma unroll
        for (int fn = 0; fn < FN; ++fn) {
            const int n = waveN * (BNt / 2) + fn * 16 + ml;
            #pragma unroll
            for (int e = 0; e < 16; ++e)
                bfr[fn][e] = Bs[n][((e >> 3) << 4) + (kb >> 1) + (e & 7)];   // 2 runs of 8 -> 2x b128
        }
        #pragma unroll
        for (int fm = 0; fm < 2; ++fm)
            #pragma unroll
            for (int fn = 0; fn < FN; ++fn)
                acc[fm][fn] = __builtin_amdgcn_wmma_f32_16x16x32_bf16(
                    false, af[fm], false, bfr[fn], (short)0, acc[fm][fn], false, false);
        __syncthreads();
    }

    // ---- epilogue: bias + addend + compile-time activation ----
    #pragma unroll
    for (int fm = 0; fm < 2; ++fm) {
        #pragma unroll
        for (int fn = 0; fn < FN; ++fn) {
            const int n  = n0 + waveN * (BNt / 2) + fn * 16 + (lane & 15);
            const float bv = bias ? bias[n] : 0.f;
            #pragma unroll
            for (int j = 0; j < 8; ++j) {
                const int m = m0 + waveM * 32 + fm * 16 + ((lane >> 4) << 3) + j;
                const long long cix = offC + (long long)m * ldc + n;
                float v = acc[fm][fn][j] + bv;
                if (addend) v += addend[cix];
                if      (ACT == 1) v = 0.5f * v * (1.0f + erff(v * 0.70710678118f)); // gelu
                else if (ACT == 2) v = 1.f / (1.f + expf(-v));                        // sigmoid
                else if (ACT == 3) v = 1.f / (1.f + expf(-(v - 2.0f)));               // sigmoid(x-2)
                else if (ACT == 4) v = tanhf(v);                                      // tanh
                if (Cf) Cf[cix] = v;
                if (Cb) Cb[cix] = (__bf16)v;
                if (CbT) CbT[(long long)(m & (Bc - 1)) * ctS1 +
                             (long long)(m >> 3) * ctS2 + (long long)n * ldcT] = (__bf16)v;
            }
        }
    }
}

// --------------------------- elementwise kernels ---------------------------

// fp32 (count,K,N) -> bf16 transposed (count,N,K); output-indexed (coalesced writes)
__global__ void k_cvt_bfT(const float* __restrict__ s, __bf16* __restrict__ d,
                          int K, int N, long long total) {
    long long i = (long long)blockIdx.x * 256 + threadIdx.x;
    if (i >= total) return;
    const long long kn = (long long)K * N;
    const int layer = (int)(i / kn);
    const long long rem = i - (long long)layer * kn;
    const int n = (int)(rem / K);
    const int k = (int)(rem - (long long)n * K);
    d[i] = (__bf16)s[(long long)layer * kn + (long long)k * N + n];
}

// h (B,S,IN) -> hT (S*B, IN) bf16
__global__ void k_transpose_h(const float* __restrict__ h, __bf16* __restrict__ d) {
    long long i = (long long)blockIdx.x * 256 + threadIdx.x;     // over SB*IN
    int c  = (int)(i % INc);
    int sb = (int)(i / INc);
    int s = sb / Bc, b = sb % Bc;
    d[i] = (__bf16)h[((long long)b * Sc + s) * INc + c];
}

// sinusoidal PE, seq descending: pe[p,c<512]=sin((F-1-p)*inv), else cos
__global__ void k_pos(__bf16* __restrict__ d) {
    long long i = (long long)blockIdx.x * 256 + threadIdx.x;     // over F*D
    int p = (int)(i / Dc), c = (int)(i % Dc);
    int cc = (c < Dc / 2) ? c : (c - Dc / 2);
    float inv = powf(10000.f, -(2.f * cc) / (float)Dc);
    float ang = (float)(Fc - 1 - p) * inv;
    d[i] = (__bf16)((c < Dc / 2) ? sinf(ang) : cosf(ang));
}

// rowwise LayerNorm over D=1024, rows<split come from src0, rest from src1
__global__ __launch_bounds__(256)
void k_ln_rows(const float* __restrict__ src0, const float* __restrict__ src1, int splitRows,
               const float* __restrict__ gs, const float* __restrict__ gb,
               __bf16* __restrict__ dst)
{
    __shared__ float red[256];
    const int row = blockIdx.x;
    const float* src = (row < splitRows) ? (src0 + (long long)row * Dc)
                                         : (src1 + (long long)(row - splitRows) * Dc);
    float v[4], sum = 0.f, sq = 0.f;
    #pragma unroll
    for (int j = 0; j < 4; ++j) {
        v[j] = src[threadIdx.x + j * 256];
        sum += v[j]; sq += v[j] * v[j];
    }
    red[threadIdx.x] = sum; __syncthreads();
    for (int o = 128; o > 0; o >>= 1) { if (threadIdx.x < o) red[threadIdx.x] += red[threadIdx.x + o]; __syncthreads(); }
    sum = red[0]; __syncthreads();
    red[threadIdx.x] = sq; __syncthreads();
    for (int o = 128; o > 0; o >>= 1) { if (threadIdx.x < o) red[threadIdx.x] += red[threadIdx.x + o]; __syncthreads(); }
    sq = red[0]; __syncthreads();
    const float mu  = sum * (1.f / Dc);
    const float var = sq * (1.f / Dc) - mu * mu;
    const float inv = rsqrtf(var + 1e-5f);
    #pragma unroll
    for (int j = 0; j < 4; ++j) {
        int c = threadIdx.x + j * 256;
        dst[(long long)row * Dc + c] = (__bf16)((v[j] - mu) * inv * gs[c] + gb[c]);
    }
}

// qu = q + u, qv = q + vb (u, vb broadcast over rows; flat over D)
__global__ void k_add_uv(const float* __restrict__ q, const float* __restrict__ u,
                         const float* __restrict__ vbp,
                         __bf16* __restrict__ qu, __bf16* __restrict__ qv) {
    long long i = (long long)blockIdx.x * 256 + threadIdx.x;     // over SB*D
    int d = (int)(i % Dc);
    float x = q[i];
    qu[i] = (__bf16)(x + u[d]);
    qv[i] = (__bf16)(x + vbp[d]);
}

// Transformer-XL rel-shift + causal mask + softmax -> bf16 probabilities
__global__ __launch_bounds__(256)
void k_shift_softmax(const float* __restrict__ cont, const float* __restrict__ posb,
                     __bf16* __restrict__ a)
{
    __shared__ float red[256];
    const int rowid = blockIdx.x;                 // (b*H+h)*S + i
    const int i = rowid & (Sc - 1);
    const long long rbase = (long long)rowid * Fc;
    const long long pbase = (long long)(rowid >> 9) * Sc * Fc;   // (b,h) slice
    float v[4];
    float mx = -3.4e38f;
    #pragma unroll
    for (int j = 0; j < 4; ++j) {
        const int jj = threadIdx.x + j * 256;
        float val = -1e30f;
        if (jj <= i + Mc) {                        // mask: j > i+M => -inf
            const int tt = i * Fc + jj + Sc;       // rel-shift flat index
            const int i2 = tt / (Fc + 1);
            const int j2 = tt - i2 * (Fc + 1);
            const float pv = (j2 == 0) ? 0.f : posb[pbase + (long long)i2 * Fc + (j2 - 1)];
            val = (cont[rbase + jj] + pv) * 0.125f;   // HD^-0.5
        }
        v[j] = val;
        mx = fmaxf(mx, val);
    }
    red[threadIdx.x] = mx; __syncthreads();
    for (int o = 128; o > 0; o >>= 1) { if (threadIdx.x < o) red[threadIdx.x] = fmaxf(red[threadIdx.x], red[threadIdx.x + o]); __syncthreads(); }
    mx = red[0]; __syncthreads();
    float sum = 0.f;
    #pragma unroll
    for (int j = 0; j < 4; ++j) { v[j] = expf(v[j] - mx); sum += v[j]; }
    red[threadIdx.x] = sum; __syncthreads();
    for (int o = 128; o > 0; o >>= 1) { if (threadIdx.x < o) red[threadIdx.x] += red[threadIdx.x + o]; __syncthreads(); }
    const float inv = 1.f / red[0];
    __syncthreads();
    #pragma unroll
    for (int j = 0; j < 4; ++j)
        a[rbase + threadIdx.x + j * 256] = (__bf16)(v[j] * inv);
}

__global__ void k_mul_bf(const float* __restrict__ r, const float* __restrict__ x,
                         __bf16* __restrict__ o) {
    long long i = (long long)blockIdx.x * 256 + threadIdx.x;
    o[i] = (__bf16)(r[i] * x[i]);
}

// out = (1-z)*x + z*h, dual fp32/bf16 outputs
__global__ void k_gru_combine(const float* __restrict__ x, const float* __restrict__ z,
                              const float* __restrict__ hc,
                              float* __restrict__ of, __bf16* __restrict__ ob) {
    long long i = (long long)blockIdx.x * 256 + threadIdx.x;
    const float zz = z[i];
    const float v = (1.f - zz) * x[i] + zz * hc[i];
    of[i] = v;
    ob[i] = (__bf16)v;
}

// (S,B,D) -> (B,S,D)
__global__ void k_transpose_out(const float* __restrict__ src, float* __restrict__ dst) {
    long long i = (long long)blockIdx.x * 256 + threadIdx.x;     // over SB*D
    int d  = (int)(i % Dc);
    int sb = (int)(i / Dc);
    int s = sb / Bc, b = sb % Bc;
    dst[((long long)b * Sc + s) * Dc + d] = src[i];
}

// ------------------------------- host side ---------------------------------

template <int ACT>
static void launch_gemm_act(hipStream_t st,
    const __bf16* A, long long aS1, long long aS2, int lda,
    const __bf16* Bm, long long bS1, long long bS2, int ldbN,
    float* Cf, __bf16* Cb, long long cS1, long long cS2, int ldc,
    const float* bias, const float* addend,
    int Md, int Nd, int Kd, int batches, int bdiv,
    __bf16* CbT, long long ctS1, long long ctS2, int ldcT)
{
    if (Nd % 128 == 0) {
        dim3 grid(Nd / 128, Md / BM, batches);
        k_gemm<128, ACT><<<grid, dim3(256), 0, st>>>(A, aS1, aS2, lda, Bm, bS1, bS2, ldbN,
                                                     Cf, Cb, CbT, ctS1, ctS2, ldcT,
                                                     cS1, cS2, ldc, bias, addend, Kd, bdiv);
    } else {
        dim3 grid(Nd / 64, Md / BM, batches);
        k_gemm<64, ACT><<<grid, dim3(256), 0, st>>>(A, aS1, aS2, lda, Bm, bS1, bS2, ldbN,
                                                    Cf, Cb, CbT, ctS1, ctS2, ldcT,
                                                    cS1, cS2, ldc, bias, addend, Kd, bdiv);
    }
}

static void launch_gemm(hipStream_t st,
    const __bf16* A, long long aS1, long long aS2, int lda,
    const __bf16* Bm, long long bS1, long long bS2, int ldbN,
    float* Cf, __bf16* Cb, long long cS1, long long cS2, int ldc,
    const float* bias, const float* addend,
    int Md, int Nd, int Kd, int batches, int bdiv, int act,
    __bf16* CbT = nullptr, long long ctS1 = 0, long long ctS2 = 0, int ldcT = 0)
{
    switch (act) {
    case 1: launch_gemm_act<1>(st, A, aS1, aS2, lda, Bm, bS1, bS2, ldbN, Cf, Cb, cS1, cS2, ldc, bias, addend, Md, Nd, Kd, batches, bdiv, CbT, ctS1, ctS2, ldcT); break;
    case 2: launch_gemm_act<2>(st, A, aS1, aS2, lda, Bm, bS1, bS2, ldbN, Cf, Cb, cS1, cS2, ldc, bias, addend, Md, Nd, Kd, batches, bdiv, CbT, ctS1, ctS2, ldcT); break;
    case 3: launch_gemm_act<3>(st, A, aS1, aS2, lda, Bm, bS1, bS2, ldbN, Cf, Cb, cS1, cS2, ldc, bias, addend, Md, Nd, Kd, batches, bdiv, CbT, ctS1, ctS2, ldcT); break;
    case 4: launch_gemm_act<4>(st, A, aS1, aS2, lda, Bm, bS1, bS2, ldbN, Cf, Cb, cS1, cS2, ldc, bias, addend, Md, Nd, Kd, batches, bdiv, CbT, ctS1, ctS2, ldcT); break;
    default: launch_gemm_act<0>(st, A, aS1, aS2, lda, Bm, bS1, bS2, ldbN, Cf, Cb, cS1, cS2, ldc, bias, addend, Md, Nd, Kd, batches, bdiv, CbT, ctS1, ctS2, ldcT); break;
    }
}

extern "C" void kernel_launch(void* const* d_in, const int* in_sizes, int n_in,
                              void* d_out, int out_size, void* d_ws, size_t ws_size,
                              hipStream_t stream)
{
    (void)in_sizes; (void)n_in; (void)out_size; (void)ws_size;

    const float* h     = (const float*)d_in[0];
    const float* mem   = (const float*)d_in[1];
    const float* We    = (const float*)d_in[2];
    const float* be    = (const float*)d_in[3];
    const float* ln1_s = (const float*)d_in[4];
    const float* ln1_b = (const float*)d_in[5];
    const float* ln2_s = (const float*)d_in[6];
    const float* ln2_b = (const float*)d_in[7];
    const float* Wq = (const float*)d_in[8];  const float* bq = (const float*)d_in[9];
    const float* Wk = (const float*)d_in[10]; const float* bk = (const float*)d_in[11];
    const float* Wv = (const float*)d_in[12]; const float* bv = (const float*)d_in[13];
    const float* Wr = (const float*)d_in[14]; const float* br = (const float*)d_in[15];
    const float* Wo = (const float*)d_in[16]; const float* bo = (const float*)d_in[17];
    const float* u  = (const float*)d_in[18]; const float* vbp = (const float*)d_in[19];
    const float* gWr = (const float*)d_in[20]; const float* gUr = (const float*)d_in[21];
    const float* gWz = (const float*)d_in[22]; const float* gUz = (const float*)d_in[23];
    const float* gWg = (const float*)d_in[24]; const float* gUg = (const float*)d_in[25];
    const float* fW1 = (const float*)d_in[26]; const float* fb1 = (const float*)d_in[27];
    const float* fW2 = (const float*)d_in[28]; const float* fb2 = (const float*)d_in[29];

    const long long DD   = (long long)Dc * Dc;            // 1M
    const long long LDD  = (long long)Lc * DD;            // 4M
    const long long L2DD = 2 * LDD;                       // 8M
    const long long LD4D = (long long)Lc * Dc * 4 * Dc;   // 16M
    const long long BHSF = (long long)Bc * Hc * Sc * Fc;  // 64M

    // ---- carve workspace (256B aligned) ----
    char* p = (char*)d_ws;
    auto take = [&](size_t bytes) -> char* {
        char* r = p; p += (bytes + 255) & ~(size_t)255; return r;
    };
    // transposed bf16 weights (N x K per matrix)
    __bf16* We_b  = (__bf16*)take((size_t)INc * Dc * 2);
    __bf16* Wq_b  = (__bf16*)take((size_t)LDD * 2);
    __bf16* Wk_b  = (__bf16*)take((size_t)LDD * 2);
    __bf16* Wv_b  = (__bf16*)take((size_t)LDD * 2);
    __bf16* Wr_b  = (__bf16*)take((size_t)LDD * 2);
    __bf16* Wo_b  = (__bf16*)take((size_t)LDD * 2);
    __bf16* gWr_b = (__bf16*)take((size_t)L2DD * 2);
    __bf16* gUr_b = (__bf16*)take((size_t)L2DD * 2);
    __bf16* gWz_b = (__bf16*)take((size_t)L2DD * 2);
    __bf16* gUz_b = (__bf16*)take((size_t)L2DD * 2);
    __bf16* gWg_b = (__bf16*)take((size_t)L2DD * 2);
    __bf16* gUg_b = (__bf16*)take((size_t)L2DD * 2);
    __bf16* fW1_b = (__bf16*)take((size_t)LD4D * 2);
    __bf16* fW2_b = (__bf16*)take((size_t)LD4D * 2);

    __bf16* hT_b   = (__bf16*)take((size_t)SBc * INc * 2);
    __bf16* pos_b  = (__bf16*)take((size_t)Fc * Dc * 2);
    float*  out_f  = (float*) take((size_t)SBc * Dc * 4);
    __bf16* out_b  = (__bf16*)take((size_t)SBc * Dc * 2);
    __bf16* kn_b   = (__bf16*)take((size_t)FBc * Dc * 2);
    float*  q_f    = (float*) take((size_t)SBc * Dc * 4);
    __bf16* qu_b   = (__bf16*)take((size_t)SBc * Dc * 2);
    __bf16* qv_b   = (__bf16*)take((size_t)SBc * Dc * 2);
    __bf16* kk_b   = (__bf16*)take((size_t)FBc * Dc * 2);
    __bf16* vT_b   = (__bf16*)take((size_t)Bc * Dc * Fc * 2);   // V^T: [b][d][f]
    __bf16* r_b    = (__bf16*)take((size_t)Fc * Dc * 2);
    float*  cont_f = (float*) take((size_t)BHSF * 4);
    float*  posb_f = (float*) take((size_t)BHSF * 4);
    __bf16* a_b    = (__bf16*)take((size_t)BHSF * 2);
    __bf16* y_b    = (__bf16*)take((size_t)SBc * Dc * 2);
    __bf16* yo_b   = (__bf16*)take((size_t)SBc * Dc * 2);
    float*  tmp1   = (float*) take((size_t)SBc * Dc * 4);
    float*  rbuf   = (float*) take((size_t)SBc * Dc * 4);
    float*  zbuf   = (float*) take((size_t)SBc * Dc * 4);
    float*  hbuf   = (float*) take((size_t)SBc * Dc * 4);
    __bf16* rx_b   = (__bf16*)take((size_t)SBc * Dc * 2);
    float*  out1_f = (float*) take((size_t)SBc * Dc * 4);
    __bf16* out1_b = (__bf16*)take((size_t)SBc * Dc * 2);
    __bf16* e_b    = (__bf16*)take((size_t)SBc * Dc * 2);
    __bf16* e1_b   = (__bf16*)take((size_t)SBc * 4 * Dc * 2);
    __bf16* e2_b   = (__bf16*)take((size_t)SBc * Dc * 2);

    auto cvtT = [&](const float* s, __bf16* d, int K, int N, long long total) {
        k_cvt_bfT<<<dim3((unsigned)((total + 255) / 256)), dim3(256), 0, stream>>>(s, d, K, N, total);
    };

    // ---- one-time weight pack: fp32 (K,N) -> bf16 transposed (N,K) ----
    cvtT(We, We_b, INc, Dc, (long long)INc * Dc);
    cvtT(Wq, Wq_b, Dc, Dc, LDD);  cvtT(Wk, Wk_b, Dc, Dc, LDD);
    cvtT(Wv, Wv_b, Dc, Dc, LDD);  cvtT(Wr, Wr_b, Dc, Dc, LDD);
    cvtT(Wo, Wo_b, Dc, Dc, LDD);
    cvtT(gWr, gWr_b, Dc, Dc, L2DD); cvtT(gUr, gUr_b, Dc, Dc, L2DD);
    cvtT(gWz, gWz_b, Dc, Dc, L2DD); cvtT(gUz, gUz_b, Dc, Dc, L2DD);
    cvtT(gWg, gWg_b, Dc, Dc, L2DD); cvtT(gUg, gUg_b, Dc, Dc, L2DD);
    cvtT(fW1, fW1_b, Dc, 4 * Dc, LD4D);
    cvtT(fW2, fW2_b, 4 * Dc, Dc, LD4D);

    const unsigned G_SBD = (unsigned)((long long)SBc * Dc / 256);

    // ---- embedding: x = gelu(hT @ We + be) ----
    k_transpose_h<<<dim3((unsigned)((long long)SBc * INc / 256)), dim3(256), 0, stream>>>(h, hT_b);
    k_pos<<<dim3((unsigned)((long long)Fc * Dc / 256)), dim3(256), 0, stream>>>(pos_b);
    launch_gemm(stream, hT_b, 0, 0, INc, We_b, 0, 0, INc,
                out_f, out_b, 0, 0, Dc, be, nullptr, SBc, Dc, INc, 1, 1, /*gelu*/1);

    // ---- GRU helper: 6 GEMMs + 2 elementwise ----
    auto gru = [&](const float* xf, const __bf16* xb, const __bf16* yb, int gidx,
                   float* of, __bf16* ob) {
        const long long o = (long long)gidx * DD;
        launch_gemm(stream, yb, 0, 0, Dc, gWr_b + o, 0, 0, Dc,
                    tmp1, nullptr, 0, 0, Dc, nullptr, nullptr, SBc, Dc, Dc, 1, 1, 0);
        launch_gemm(stream, xb, 0, 0, Dc, gUr_b + o, 0, 0, Dc,
                    rbuf, nullptr, 0, 0, Dc, nullptr, tmp1, SBc, Dc, Dc, 1, 1, /*sigmoid*/2);
        launch_gemm(stream, yb, 0, 0, Dc, gWz_b + o, 0, 0, Dc,
                    tmp1, nullptr, 0, 0, Dc, nullptr, nullptr, SBc, Dc, Dc, 1, 1, 0);
        launch_gemm(stream, xb, 0, 0, Dc, gUz_b + o, 0, 0, Dc,
                    zbuf, nullptr, 0, 0, Dc, nullptr, tmp1, SBc, Dc, Dc, 1, 1, /*sig-2*/3);
        launch_gemm(stream, yb, 0, 0, Dc, gWg_b + o, 0, 0, Dc,
                    tmp1, nullptr, 0, 0, Dc, nullptr, nullptr, SBc, Dc, Dc, 1, 1, 0);
        k_mul_bf<<<G_SBD, 256, 0, stream>>>(rbuf, xf, rx_b);
        launch_gemm(stream, rx_b, 0, 0, Dc, gUg_b + o, 0, 0, Dc,
                    hbuf, nullptr, 0, 0, Dc, nullptr, tmp1, SBc, Dc, Dc, 1, 1, /*tanh*/4);
        k_gru_combine<<<G_SBD, 256, 0, stream>>>(xf, zbuf, hbuf, of, ob);
    };

    for (int i = 0; i < Lc; ++i) {
        const float* mem_i = mem + (long long)i * Mc * Bc * Dc;

        // LN over concat([memory, out]); q-norm = tail SB rows of k-norm
        k_ln_rows<<<FBc, 256, 0, stream>>>(mem_i, out_f, Mc * Bc,
                                           ln1_s + i * Dc, ln1_b + i * Dc, kn_b);
        // q/k/v/r projections
        launch_gemm(stream, kn_b + (long long)Mc * Bc * Dc, 0, 0, Dc,
                    Wq_b + (long long)i * DD, 0, 0, Dc,
                    q_f, nullptr, 0, 0, Dc, bq + i * Dc, nullptr, SBc, Dc, Dc, 1, 1, 0);
        launch_gemm(stream, kn_b, 0, 0, Dc, Wk_b + (long long)i * DD, 0, 0, Dc,
                    nullptr, kk_b, 0, 0, Dc, bk + i * Dc, nullptr, FBc, Dc, Dc, 1, 1, 0);
        // v projection: write V^T[b][d][f] directly (k-contiguous for A@V)
        launch_gemm(stream, kn_b, 0, 0, Dc, Wv_b + (long long)i * DD, 0, 0, Dc,
                    nullptr, nullptr, 0, 0, Dc, bv + i * Dc, nullptr, FBc, Dc, Dc, 1, 1, 0,
                    vT_b, (long long)Dc * Fc, 1, Fc);
        launch_gemm(stream, pos_b, 0, 0, Dc, Wr_b + (long long)i * DD, 0, 0, Dc,
                    nullptr, r_b, 0, 0, Dc, br + i * Dc, nullptr, Fc, Dc, Dc, 1, 1, 0);
        k_add_uv<<<G_SBD, 256, 0, stream>>>(q_f, u + i * Dc, vbp + i * Dc, qu_b, qv_b);

        // content = (q+u) @ K^T, batched over (b,h); B = kk_b rows (k-contig)
        launch_gemm(stream, qu_b, Dc, HDc, Bc * Dc,
                    kk_b, Dc, HDc, Bc * Dc,
                    cont_f, nullptr, (long long)Hc * Sc * Fc, (long long)Sc * Fc, Fc,
                    nullptr, nullptr, Sc, Fc, HDc, Bc * Hc, Hc, 0);
        // position = (q+vb) @ R^T (R shared over batch)
        launch_gemm(stream, qv_b, Dc, HDc, Bc * Dc,
                    r_b, 0, HDc, Dc,
                    posb_f, nullptr, (long long)Hc * Sc * Fc, (long long)Sc * Fc, Fc,
                    nullptr, nullptr, Sc, Fc, HDc, Bc * Hc, Hc, 0);
        // rel-shift + mask + softmax -> bf16 probs
        k_shift_softmax<<<Bc * Hc * Sc, 256, 0, stream>>>(cont_f, posb_f, a_b);
        // y = A @ V   (B = V^T slice, k-contiguous; N = 64 -> BNt=64)
        launch_gemm(stream, a_b, (long long)Hc * Sc * Fc, (long long)Sc * Fc, Fc,
                    vT_b, (long long)Dc * Fc, (long long)HDc * Fc, Fc,
                    nullptr, y_b, Dc, HDc, Bc * Dc,
                    nullptr, nullptr, Sc, HDc, Fc, Bc * Hc, Hc, 0);
        // yo = gelu(y @ Wo + bo)
        launch_gemm(stream, y_b, 0, 0, Dc, Wo_b + (long long)i * DD, 0, 0, Dc,
                    nullptr, yo_b, 0, 0, Dc, bo + i * Dc, nullptr, SBc, Dc, Dc, 1, 1, 1);

        // GRU 1: out1 = gru(out, yo)
        gru(out_f, out_b, yo_b, i * 2 + 0, out1_f, out1_b);

        // FFN: e2 = gelu(gelu(LN(out1) @ fW1 + fb1) @ fW2 + fb2)
        k_ln_rows<<<SBc, 256, 0, stream>>>(out1_f, out1_f, SBc,
                                           ln2_s + i * Dc, ln2_b + i * Dc, e_b);
        launch_gemm(stream, e_b, 0, 0, Dc,
                    fW1_b + (long long)i * Dc * 4 * Dc, 0, 0, Dc,
                    nullptr, e1_b, 0, 0, 4 * Dc, fb1 + i * 4 * Dc, nullptr,
                    SBc, 4 * Dc, Dc, 1, 1, 1);
        launch_gemm(stream, e1_b, 0, 0, 4 * Dc,
                    fW2_b + (long long)i * 4 * Dc * Dc, 0, 0, 4 * Dc,
                    nullptr, e2_b, 0, 0, Dc, fb2 + i * Dc, nullptr,
                    SBc, Dc, 4 * Dc, 1, 1, 1);

        // GRU 2: out = gru(out1, e2)
        gru(out1_f, out1_b, e2_b, i * 2 + 1, out_f, out_b);
    }

    // (S,B,D) -> (B,S,D)
    k_transpose_out<<<G_SBD, 256, 0, stream>>>(out_f, (float*)d_out);
}
